// TransformerDecoder_67190468379063
// MI455X (gfx1250) — matrix operand order repaired and checked by reading, not compiled, not run on hardware
//
#include <hip/hip_runtime.h>
#include <math.h>

// ---------------------------------------------------------------------------
// GPT forward + cross-entropy for gfx1250 (MI455X).
// - Weights pre-converted once per launch: f32 -> bf16, stored transposed
//   Wt[N][K] (K-contiguous).
// - All GEMM-consumed activations are produced in bf16 -> both GEMM tiles
//   stage via global_load_async_to_lds_b128 with ping-pong double buffering.
// - GEMM core: v_wmma_f32_16x16x32_bf16, f32 accumulate, fused epilogue.
// - Attention: Q.K^T and P.V on the WMMA pipe too; online softmax on VALU.
// ---------------------------------------------------------------------------

#define Bb   4
#define Tt   1024
#define Cc   1024
#define Hh   16
#define HSs  64
#define Vv   32000
#define Ll   8
#define NT   (Bb * Tt)        // 4096 token rows
#define FF   (4 * Cc)         // 4096 MLP hidden

typedef __bf16          v16bf  __attribute__((ext_vector_type(16)));
typedef float           v8f    __attribute__((ext_vector_type(8)));
typedef unsigned short  u16x4  __attribute__((ext_vector_type(4)));
typedef unsigned short  u16x8  __attribute__((ext_vector_type(8)));
typedef unsigned short  u16x16 __attribute__((ext_vector_type(16)));

static __device__ __forceinline__ unsigned short f32_to_bf16_rne(float f) {
    unsigned u = __float_as_uint(f);
    u += 0x7FFFu + ((u >> 16) & 1u);          // round-to-nearest-even
    return (unsigned short)(u >> 16);
}

// Assemble a 16-element bf16 fragment from two 16-byte LDS loads.
static __device__ __forceinline__ v16bf load_frag16(const unsigned short* p_lo,
                                                    const unsigned short* p_hi) {
    u16x8 lo = *(const u16x8*)p_lo;
    u16x8 hi = *(const u16x8*)p_hi;
    u16x16 c = __builtin_shufflevector(lo, hi, 0,1,2,3,4,5,6,7,8,9,10,11,12,13,14,15);
    return __builtin_bit_cast(v16bf, c);
}

// Async global->LDS 16B copy (per-lane). dst = LDS byte offset, src = global.
static __device__ __forceinline__ void async_g2l_b128(unsigned lds_off,
                                                      unsigned long long gaddr) {
    asm volatile("global_load_async_to_lds_b128 %0, %1, off"
                 :: "v"(lds_off), "v"(gaddr) : "memory");
}

static __device__ __forceinline__ void wait_async0() {
#if defined(__has_builtin)
#if __has_builtin(__builtin_amdgcn_s_wait_asynccnt)
    __builtin_amdgcn_s_wait_asynccnt(0);
    return;
#endif
#endif
    asm volatile("s_wait_asynccnt 0x0" ::: "memory");
}

static __device__ __forceinline__ void wait_async4() {
#if defined(__has_builtin)
#if __has_builtin(__builtin_amdgcn_s_wait_asynccnt)
    __builtin_amdgcn_s_wait_asynccnt(4);
    return;
#endif
#endif
    asm volatile("s_wait_asynccnt 0x4" ::: "memory");
}

// ---------------------------------------------------------------------------
// Weight pre-pass: Wt[n][k] = bf16(W[k][n]).  Grid: (N/32, K/32), 256 thr.
// ---------------------------------------------------------------------------
__global__ __launch_bounds__(256)
void k_cvt_transpose(const float* __restrict__ W, unsigned short* __restrict__ Wt,
                     int K, int N)
{
    __shared__ unsigned short tile[32][33];
    const int tx = threadIdx.x & 31, ty = threadIdx.x >> 5;   // 32 x 8
    const int nb = blockIdx.x * 32, kb = blockIdx.y * 32;
    #pragma unroll
    for (int i = 0; i < 4; ++i) {
        int k = kb + ty + i * 8;
        tile[tx][ty + i * 8] = f32_to_bf16_rne(W[(size_t)k * N + nb + tx]);
    }
    __syncthreads();
    #pragma unroll
    for (int i = 0; i < 4; ++i) {
        int n = nb + ty + i * 8;
        Wt[(size_t)n * K + kb + tx] = tile[ty + i * 8][tx];
    }
}

// ---------------------------------------------------------------------------
// WMMA GEMM: Y = relu?( X[M,K](bf16) @ Wt[N,K](bf16)^T + bias[N] + res[M,N] )
// Output to Yf (f32) or Yh (bf16) -- exactly one is non-null.
// Block tile 128x128, K-step 32, ping-pong LDS, 8 waves, wave tile 64x32.
// Requires M%128==0, N%128==0, K%32==0 (true for all call sites here).
// ---------------------------------------------------------------------------
__global__ __launch_bounds__(256)
void k_gemm_bf16(const unsigned short* __restrict__ X,
                 const unsigned short* __restrict__ Wt,
                 const float* __restrict__ bias, const float* __restrict__ res,
                 float* __restrict__ Yf, unsigned short* __restrict__ Yh,
                 int M, int N, int K, int relu)
{
    // stride 40 ushorts (80B) keeps every 16B fragment load aligned & de-conflicted
    __shared__ unsigned short As[2][128][40];   // [buf][m][k] bf16
    __shared__ unsigned short Bs[2][128][40];   // [buf][n][k] bf16

    const int t    = threadIdx.x;
    const int m0   = blockIdx.y * 128;
    const int n0   = blockIdx.x * 128;
    const int wave = t >> 5, lane = t & 31;
    const int wm   = wave >> 2, wn = wave & 3;        // 2x4 wave grid
    const int lr   = lane & 15, lh = lane >> 4;

    const int cr = t >> 2, cq = t & 3;                // staging row / 16B quarter
    unsigned la0[2], la1[2], lb0[2], lb1[2];
    #pragma unroll
    for (int bf = 0; bf < 2; ++bf) {
        la0[bf] = (unsigned)(size_t)&As[bf][cr][cq * 8];
        la1[bf] = (unsigned)(size_t)&As[bf][64 + cr][cq * 8];
        lb0[bf] = (unsigned)(size_t)&Bs[bf][cr][cq * 8];
        lb1[bf] = (unsigned)(size_t)&Bs[bf][64 + cr][cq * 8];
    }
    const unsigned short* xr0 = X  + (size_t)(m0 + cr)      * K + cq * 8;
    const unsigned short* xr1 = X  + (size_t)(m0 + 64 + cr) * K + cq * 8;
    const unsigned short* wr0 = Wt + (size_t)(n0 + cr)      * K + cq * 8;
    const unsigned short* wr1 = Wt + (size_t)(n0 + 64 + cr) * K + cq * 8;

    v8f acc[4][2] = {};                               // 4 M-tiles x 2 N-tiles

    // prologue: stage tile 0 into buffer 0
    async_g2l_b128(la0[0], (unsigned long long)(size_t)(xr0));
    async_g2l_b128(la1[0], (unsigned long long)(size_t)(xr1));
    async_g2l_b128(lb0[0], (unsigned long long)(size_t)(wr0));
    async_g2l_b128(lb1[0], (unsigned long long)(size_t)(wr1));

    for (int k0 = 0, it = 0; k0 < K; k0 += 32, ++it) {
        const int cur = it & 1, nxt = cur ^ 1;
        const bool havenext = (k0 + 32 < K);
        if (havenext) {                               // stage tile i+1 under compute
            async_g2l_b128(la0[nxt], (unsigned long long)(size_t)(xr0 + k0 + 32));
            async_g2l_b128(la1[nxt], (unsigned long long)(size_t)(xr1 + k0 + 32));
            async_g2l_b128(lb0[nxt], (unsigned long long)(size_t)(wr0 + k0 + 32));
            async_g2l_b128(lb1[nxt], (unsigned long long)(size_t)(wr1 + k0 + 32));
            if (k0 + 64 < K)
                __builtin_prefetch(Wt + (size_t)(n0 + (t >> 1)) * K + k0 + 64 + (t & 1) * 16, 0, 3);
            wait_async4();                            // tile i complete (in-order)
        } else {
            wait_async0();
        }
        __syncthreads();

        // ---- fragments per ISA VGPR layouts ----
        const unsigned short (*Ac)[40] = As[cur];
        const unsigned short (*Bc)[40] = Bs[cur];
        v16bf af[4];
        #pragma unroll
        for (int mi = 0; mi < 4; ++mi) {
            const unsigned short* r = Ac[wm * 64 + mi * 16 + lr];
            af[mi] = load_frag16(&r[lh * 8], &r[lh * 8 + 16]);
        }
        v16bf bfr[2];
        #pragma unroll
        for (int ni = 0; ni < 2; ++ni) {
            const unsigned short* r = Bc[wn * 32 + ni * 16 + lr];
            bfr[ni] = load_frag16(&r[lh * 8], &r[lh * 8 + 16]);
        }
        #pragma unroll
        for (int mi = 0; mi < 4; ++mi)
            #pragma unroll
            for (int ni = 0; ni < 2; ++ni)
                acc[mi][ni] = __builtin_amdgcn_wmma_f32_16x16x32_bf16(
                    false, af[mi], false, bfr[ni], (short)0, acc[mi][ni], false, false);
        __syncthreads();
    }

    // ---- epilogue: D layout = lane lr column, rows m = lh*8 + j ----
    #pragma unroll
    for (int mi = 0; mi < 4; ++mi) {
        #pragma unroll
        for (int ni = 0; ni < 2; ++ni) {
            int n  = n0 + wn * 32 + ni * 16 + lr;
            float bv = bias ? bias[n] : 0.0f;
            int mb = m0 + wm * 64 + mi * 16 + lh * 8;
            #pragma unroll
            for (int j = 0; j < 8; ++j) {
                float v = acc[mi][ni][j] + bv;
                if (res)  v += res[(size_t)(mb + j) * N + n];
                if (relu) v = fmaxf(v, 0.0f);
                if (Yf) Yf[(size_t)(mb + j) * N + n] = v;
                else    Yh[(size_t)(mb + j) * N + n] = f32_to_bf16_rne(v);
            }
        }
    }
}

// ---------------------------------------------------------------------------
// Embedding: x[bt, c] = tok_emb[idx[bt], c] + pos_emb[bt % T, c]   (f32 out)
// ---------------------------------------------------------------------------
__global__ __launch_bounds__(256)
void k_embed(const float* __restrict__ tok, const float* __restrict__ pos,
             const int* __restrict__ idx, float* __restrict__ x)
{
    int id = blockIdx.x * 256 + threadIdx.x;          // grid sized to NT*C
    int bt = id >> 10;                                 // /C
    int c  = id & (Cc - 1);
    x[id] = tok[(size_t)idx[bt] * Cc + c] + pos[(size_t)(bt & (Tt - 1)) * Cc + c];
}

// ---------------------------------------------------------------------------
// LayerNorm over C=1024 -> bf16 output (consumed only as GEMM A-operand).
// ---------------------------------------------------------------------------
__global__ __launch_bounds__(256)
void k_layernorm(const float* __restrict__ x, const float* __restrict__ g,
                 const float* __restrict__ b, unsigned short* __restrict__ y)
{
    __shared__ float red[256];
    const int t = threadIdx.x;
    const size_t row = blockIdx.x;
    const float4 xv = *(const float4*)(x + row * Cc + t * 4);

    red[t] = xv.x + xv.y + xv.z + xv.w;
    __syncthreads();
    for (int s = 128; s > 0; s >>= 1) { if (t < s) red[t] += red[t + s]; __syncthreads(); }
    const float mean = red[0] * (1.0f / Cc);
    __syncthreads();

    float dx = xv.x - mean, dy = xv.y - mean, dz = xv.z - mean, dw = xv.w - mean;
    red[t] = dx * dx + dy * dy + dz * dz + dw * dw;
    __syncthreads();
    for (int s = 128; s > 0; s >>= 1) { if (t < s) red[t] += red[t + s]; __syncthreads(); }
    const float rstd = rsqrtf(red[0] * (1.0f / Cc) + 1e-5f);

    const float4 gv = *(const float4*)(g + t * 4);
    const float4 bv = *(const float4*)(b + t * 4);
    u16x4 o;
    o.x = f32_to_bf16_rne(dx * rstd * gv.x + bv.x);
    o.y = f32_to_bf16_rne(dy * rstd * gv.y + bv.y);
    o.z = f32_to_bf16_rne(dz * rstd * gv.z + bv.z);
    o.w = f32_to_bf16_rne(dw * rstd * gv.w + bv.w);
    *(u16x4*)(y + row * Cc + t * 4) = o;
}

// ---------------------------------------------------------------------------
// WMMA flash attention. Block = (64 queries) x (one b,h), 256 threads.
// Q/K/V are bf16 [B*T, C] (head slice at h*HS). Output bf16.
// Per key tile:  S = Q.K^T on WMMA  ->  online softmax on VALU (4 thr/query)
//             -> O = a*O + P.V on WMMA (V transposed in LDS, K-contiguous).
// Wave tiling: tm = wave&3 (query tile), tn = 2*(wave>>2)+{0,1} (col tiles).
// ---------------------------------------------------------------------------
__global__ __launch_bounds__(256)
void k_attention(const unsigned short* __restrict__ Q,
                 const unsigned short* __restrict__ K,
                 const unsigned short* __restrict__ V,
                 const int* __restrict__ idx,
                 unsigned short* __restrict__ O)
{
    __shared__ unsigned short Qh[64][72];   // [query][d]
    __shared__ unsigned short Kh[64][72];   // [key][d]
    __shared__ unsigned short Vt[64][72];   // [d][key]   (transposed)
    __shared__ unsigned short Ph[64][72];   // [query][key] probabilities
    __shared__ float Ss[64][64];            // f32 scores
    __shared__ float alphas[64];
    __shared__ float lsum[64];
    __shared__ unsigned char kpad[Tt];      // pad-mask flags for this batch

    const int t  = threadIdx.x;
    const int bh = blockIdx.y;
    const int b  = bh / Hh, h = bh % Hh;
    const int q0 = blockIdx.x * 64;
    const float scale = 0.125f;             // HS^-0.5

    const int wave = t >> 5, lane = t & 31;
    const int lr = lane & 15, lh = lane >> 4;
    const int tm = wave & 3;                // query 16-tile
    const int tn0 = (wave >> 2) * 2;        // first column 16-tile

    const int qrel = t >> 2;                // softmax mapping: 4 threads/query
    const int kseg = (t & 3) * 16;
    const int* idx_b = idx + b * Tt;
    const unsigned short* Qg = Q + ((size_t)b * Tt) * Cc + h * HSs;
    const unsigned short* Kg = K + ((size_t)b * Tt) * Cc + h * HSs;
    const unsigned short* Vg = V + ((size_t)b * Tt) * Cc + h * HSs;

    // pad flags + async-stage Q tile (64 rows x 128B = 512 16B chunks)
    for (int i = t; i < Tt; i += 256) kpad[i] = (idx_b[i] != 0);
    #pragma unroll
    for (int i = 0; i < 2; ++i) {
        int c = t + 256 * i, row = c >> 3, q8 = c & 7;
        async_g2l_b128((unsigned)(size_t)&Qh[row][q8 * 8],
                       (unsigned long long)(size_t)(Qg + (size_t)(q0 + row) * Cc + q8 * 8));
    }

    v8f accO[2] = {};
    float m = -1e30f, l = 0.0f;

    for (int kt0 = 0; kt0 < q0 + 64; kt0 += 64) {
        __syncthreads();                    // previous-tile readers done
        // stage K tile (async, K-contiguous rows)
        #pragma unroll
        for (int i = 0; i < 2; ++i) {
            int c = t + 256 * i, row = c >> 3, q8 = c & 7;
            async_g2l_b128((unsigned)(size_t)&Kh[row][q8 * 8],
                           (unsigned long long)(size_t)(Kg + (size_t)(kt0 + row) * Cc + q8 * 8));
        }
        // stage V transposed: Vt[d][key]  (16 elems/thread)
        {
            int row = t >> 2, seg = (t & 3) * 16;
            const unsigned short* src = Vg + (size_t)(kt0 + row) * Cc + seg;
            u16x8 v0 = *(const u16x8*)src;
            u16x8 v1 = *(const u16x8*)(src + 8);
            #pragma unroll
            for (int e = 0; e < 8; ++e) Vt[seg + e][row] = v0[e];
            #pragma unroll
            for (int e = 0; e < 8; ++e) Vt[seg + 8 + e][row] = v1[e];
        }
        wait_async0();
        __syncthreads();

        // ---- scores on WMMA: 2 tiles x 2 k-steps per wave ----
        v8f sacc[2] = {};
        #pragma unroll
        for (int tni = 0; tni < 2; ++tni) {
            const int tn = tn0 + tni;
            #pragma unroll
            for (int d0 = 0; d0 < HSs; d0 += 32) {
                const unsigned short* qa = Qh[tm * 16 + lr];
                const unsigned short* kb_ = Kh[tn * 16 + lr];
                v16bf a = load_frag16(&qa[d0 + lh * 8], &qa[d0 + lh * 8 + 16]);
                v16bf bb = load_frag16(&kb_[d0 + lh * 8], &kb_[d0 + lh * 8 + 16]);
                sacc[tni] = __builtin_amdgcn_wmma_f32_16x16x32_bf16(
                    false, a, false, bb, (short)0, sacc[tni], false, false);
            }
        }
        // masked, scaled scores -> Ss
        #pragma unroll
        for (int tni = 0; tni < 2; ++tni) {
            const int tn = tn0 + tni;
            const int col = tn * 16 + lr, kg = kt0 + col;
            #pragma unroll
            for (int j = 0; j < 8; ++j) {
                int row = tm * 16 + lh * 8 + j;
                bool valid = (kg <= q0 + row) && kpad[kg];
                Ss[row][col] = valid ? sacc[tni][j] * scale : -1e30f;
            }
        }
        __syncthreads();

        // ---- online softmax (4 threads per query, redundant stats) ----
        float mt = -1e30f;
        #pragma unroll 8
        for (int j = 0; j < 64; ++j) mt = fmaxf(mt, Ss[qrel][j]);
        float mn = fmaxf(m, mt);
        float alpha = __expf(m - mn);
        l *= alpha;
        #pragma unroll 4
        for (int j = 0; j < 64; ++j) {
            float p = __expf(Ss[qrel][j] - mn);
            l += p;
            if ((j >> 4) == (t & 3)) Ph[qrel][j] = f32_to_bf16_rne(p);
        }
        m = mn;
        if ((t & 3) == 0) alphas[qrel] = alpha;
        __syncthreads();

        // ---- rescale O accumulators, then O += P.V on WMMA ----
        #pragma unroll
        for (int tni = 0; tni < 2; ++tni) {
            #pragma unroll
            for (int j = 0; j < 8; ++j)
                accO[tni][j] *= alphas[tm * 16 + lh * 8 + j];
        }
        #pragma unroll
        for (int tni = 0; tni < 2; ++tni) {
            const int tn = tn0 + tni;                 // d 16-tile
            #pragma unroll
            for (int key0 = 0; key0 < 64; key0 += 32) {
                const unsigned short* pa = Ph[tm * 16 + lr];
                const unsigned short* vb_ = Vt[tn * 16 + lr];
                v16bf a = load_frag16(&pa[key0 + lh * 8], &pa[key0 + lh * 8 + 16]);
                v16bf bb = load_frag16(&vb_[key0 + lh * 8], &vb_[key0 + lh * 8 + 16]);
                accO[tni] = __builtin_amdgcn_wmma_f32_16x16x32_bf16(
                    false, a, false, bb, (short)0, accO[tni], false, false);
            }
        }
    }

    if ((t & 3) == 0) lsum[qrel] = l;
    __syncthreads();

    // ---- epilogue: O[q][d] = accO / l ----
    #pragma unroll
    for (int tni = 0; tni < 2; ++tni) {
        const int d = (tn0 + tni) * 16 + lr;
        #pragma unroll
        for (int j = 0; j < 8; ++j) {
            int row = tm * 16 + lh * 8 + j;
            float rl = 1.0f / fmaxf(lsum[row], 1e-30f);
            O[((size_t)b * Tt + q0 + row) * Cc + h * HSs + d] =
                f32_to_bf16_rne(accO[tni][j] * rl);
        }
    }
}

// ---------------------------------------------------------------------------
// Cross-entropy: per-row  -(logit[tgt] - max - log(sum exp))
// ---------------------------------------------------------------------------
__global__ __launch_bounds__(256)
void k_loss_rows(const float* __restrict__ logits, const int* __restrict__ tgt,
                 float* __restrict__ rowloss)
{
    __shared__ float red[256];
    const int t = threadIdx.x;
    const size_t row = blockIdx.x;
    const float* lp = logits + row * Vv;

    float mx = -1e30f;
    for (int i = t; i < Vv; i += 256) mx = fmaxf(mx, lp[i]);
    red[t] = mx;
    __syncthreads();
    for (int s = 128; s > 0; s >>= 1) { if (t < s) red[t] = fmaxf(red[t], red[t + s]); __syncthreads(); }
    const float rmax = red[0];
    __syncthreads();

    float sum = 0.0f;
    for (int i = t; i < Vv; i += 256) sum += __expf(lp[i] - rmax);
    red[t] = sum;
    __syncthreads();
    for (int s = 128; s > 0; s >>= 1) { if (t < s) red[t] += red[t + s]; __syncthreads(); }

    if (t == 0)
        rowloss[row] = -(lp[tgt[row]] - rmax - logf(red[0]));
}

__global__ __launch_bounds__(256)
void k_loss_reduce(const float* __restrict__ rowloss, float* __restrict__ out)
{
    __shared__ float red[256];
    const int t = threadIdx.x;
    float s = 0.0f;
    for (int i = t; i < NT; i += 256) s += rowloss[i];
    red[t] = s;
    __syncthreads();
    for (int st = 128; st > 0; st >>= 1) { if (t < st) red[t] += red[t + st]; __syncthreads(); }
    if (t == 0) out[0] = red[0] * (1.0f / NT);
}

// ---------------------------------------------------------------------------
// Host orchestration.
// Input order (setup_inputs dict flattened): tok_emb, pos_emb,
//   8 x {wq,wk,wv,wo,bo,ln1_g,ln1_b,ln2_g,ln2_b,w1,b1,w2,b2},
//   lnf_g, lnf_b, lm_w, lm_b, idx, targets.
// d_out: logits (B*T*V f32) then loss (1 f32).
// ---------------------------------------------------------------------------
static void run_gemm(const unsigned short* X, const float* Wf32,
                     unsigned short* wt_scratch, const float* bias,
                     const float* res, float* Yf, unsigned short* Yh,
                     int M, int N, int K, int relu, hipStream_t s)
{
    dim3 tgrid(N / 32, K / 32);
    k_cvt_transpose<<<tgrid, 256, 0, s>>>(Wf32, wt_scratch, K, N);
    dim3 grid(N / 128, M / 128);
    k_gemm_bf16<<<grid, 256, 0, s>>>(X, wt_scratch, bias, res, Yf, Yh, M, N, K, relu);
}

extern "C" void kernel_launch(void* const* d_in, const int* in_sizes, int n_in,
                              void* d_out, int out_size, void* d_ws, size_t ws_size,
                              hipStream_t stream)
{
    (void)in_sizes; (void)n_in; (void)out_size; (void)ws_size;

    const float* tok_emb = (const float*)d_in[0];
    const float* pos_emb = (const float*)d_in[1];
    const float* lnf_g   = (const float*)d_in[2 + 13 * Ll + 0];
    const float* lnf_b   = (const float*)d_in[2 + 13 * Ll + 1];
    const float* lm_w    = (const float*)d_in[2 + 13 * Ll + 2];
    const float* lm_b    = (const float*)d_in[2 + 13 * Ll + 3];
    const int*   idx     = (const int*)  d_in[2 + 13 * Ll + 4];
    const int*   targets = (const int*)  d_in[2 + 13 * Ll + 5];

    // workspace carve-up
    char* wsb = (char*)d_ws;
    const size_t Af = (size_t)NT * Cc * sizeof(float);
    const size_t Ah = (size_t)NT * Cc * sizeof(unsigned short);
    float* x_a   = (float*)wsb;                wsb += Af;
    float* x_b   = (float*)wsb;                wsb += Af;
    float* rloss = (float*)wsb;                wsb += (size_t)NT * sizeof(float);
    unsigned short* hbuf = (unsigned short*)wsb;  wsb += Ah;   // LN out (bf16)
    unsigned short* qb   = (unsigned short*)wsb;  wsb += Ah;   // q (bf16)
    unsigned short* kb   = (unsigned short*)wsb;  wsb += Ah;   // k (bf16)
    unsigned short* vb   = (unsigned short*)wsb;  wsb += Ah;   // v (bf16)
    unsigned short* ob   = (unsigned short*)wsb;  wsb += Ah;   // attn out (bf16)
    unsigned short* mlp  = (unsigned short*)wsb;  wsb += (size_t)NT * FF * sizeof(unsigned short);
    unsigned short* wt   = (unsigned short*)wsb;  // bf16 weight scratch (<= C*V = 65.6MB)

    float* logits = (float*)d_out;
    float* loss   = logits + (size_t)NT * Vv;

    // embedding
    k_embed<<<(NT * Cc) / 256, 256, 0, stream>>>(tok_emb, pos_emb, idx, x_a);

    for (int l = 0; l < Ll; ++l) {
        const float* const* p = (const float* const*)(d_in + 2 + 13 * l);
        const float *wq = p[0], *wk = p[1], *wv = p[2], *wo = p[3], *bo = p[4];
        const float *g1 = p[5], *b1g = p[6], *g2 = p[7], *b2g = p[8];
        const float *w1 = p[9], *bb1 = p[10], *w2 = p[11], *bb2 = p[12];

        k_layernorm<<<NT, 256, 0, stream>>>(x_a, g1, b1g, hbuf);
        run_gemm(hbuf, wq, wt, nullptr, nullptr, nullptr, qb, NT, Cc, Cc, 0, stream);
        run_gemm(hbuf, wk, wt, nullptr, nullptr, nullptr, kb, NT, Cc, Cc, 0, stream);
        run_gemm(hbuf, wv, wt, nullptr, nullptr, nullptr, vb, NT, Cc, Cc, 0, stream);

        dim3 agrid(Tt / 64, Bb * Hh);
        k_attention<<<agrid, 256, 0, stream>>>(qb, kb, vb, idx, ob);

        run_gemm(ob, wo, wt, bo, x_a, x_b, nullptr, NT, Cc, Cc, 0, stream);   // x_b = x_a + o@wo + bo
        k_layernorm<<<NT, 256, 0, stream>>>(x_b, g2, b2g, hbuf);
        run_gemm(hbuf, w1, wt, bb1, nullptr, nullptr, mlp, NT, FF, Cc, 1, stream); // relu, bf16 out
        run_gemm(mlp, w2, wt, bb2, x_b, x_a, nullptr, NT, Cc, FF, 0, stream);      // x_a = x_b + ...
    }

    k_layernorm<<<NT, 256, 0, stream>>>(x_a, lnf_g, lnf_b, hbuf);
    run_gemm(hbuf, lm_w, wt, lm_b, nullptr, logits, nullptr, NT, Vv, Cc, 0, stream);

    k_loss_rows<<<NT, 256, 0, stream>>>(logits, targets, rloss);
    k_loss_reduce<<<1, 256, 0, stream>>>(rloss, loss);
}
